// AudioMambaModel_72464688218614
// MI455X (gfx1250) — matrix operand order repaired and checked by reading, not compiled, not run on hardware
//
#include <hip/hip_runtime.h>
#include <hip/hip_bf16.h>
#include <math.h>

// ---------------- problem constants ----------------
static constexpr int BATCH   = 8;
static constexpr int LSEQ    = 8192;
static constexpr int DMODEL  = 256;
static constexpr int DINNER  = 512;
static constexpr int M_TOK   = BATCH * LSEQ;   // 65536 tokens
static constexpr int DTRANK  = 16;
static constexpr int DSTATE  = 4;
static constexpr int PROJ_LD = 32;             // padded (dt_rank + 2*d_state = 24 -> 32)

// ---------------- vector types for WMMA ----------------
typedef __bf16 v16bf __attribute__((ext_vector_type(16)));
typedef __bf16 v8bf  __attribute__((ext_vector_type(8)));
typedef float  v8f   __attribute__((ext_vector_type(8)));

__device__ __forceinline__ __bf16 f2bf(float f) {
    unsigned u = __builtin_bit_cast(unsigned, f);
    unsigned r = (u + 0x7FFFu + ((u >> 16) & 1u)) >> 16;
    return __builtin_bit_cast(__bf16, (unsigned short)r);
}

__device__ __forceinline__ float silu_f(float v) { return v / (1.0f + __expf(-v)); }

// ---------------- weight convert + zero-pad to bf16 (N,K) row-major ----------------
__global__ void convert_pad_kernel(const float* __restrict__ src, __bf16* __restrict__ dst,
                                   int rows_src, int rows_dst, int cols_src, int cols_dst) {
    int i = blockIdx.x * blockDim.x + threadIdx.x;
    int total = rows_dst * cols_dst;
    if (i >= total) return;
    int r = i / cols_dst, c = i % cols_dst;
    float v = (r < rows_src && c < cols_src) ? src[r * cols_src + c] : 0.0f;
    dst[i] = f2bf(v);
}

// ---------------- generic WMMA GEMM:  C[m,n] = sum_k A[m,k] * W[n,k] ----------------
// A: f32, row stride lda, logical K with zero-fill for k >= Kreal (staged to LDS as bf16)
// W: bf16, (N,K) row-major (pre-padded). C: f32, row stride ldc.
// block = 128 threads (4 waves); wave w handles N-tile (blockIdx.x*4 + w); blockIdx.y = M-tile.
// epilogue: 0 = plain store; 1 = softplus(v + bias[n])
__global__ void gemm_bf16_wmma(const float* __restrict__ A, int lda, int Kreal,
                               const __bf16* __restrict__ W, float* __restrict__ C,
                               int ldc, int N, int K, int epilogue,
                               const float* __restrict__ bias) {
    extern __shared__ char smem_raw[];
    __bf16* sA = (__bf16*)smem_raw;                 // 16 x K, row-major
    const int tid = threadIdx.x;

    // cooperative stage: A tile (16 rows x K) -> LDS, f32 -> bf16, zero-fill k >= Kreal
    const float* Ablk = A + (size_t)blockIdx.y * 16 * lda;
    const int total = 16 * K;
    for (int i = tid; i < total; i += blockDim.x) {
        int r = i / K, k = i - r * K;
        float v = (k < Kreal) ? Ablk[(size_t)r * lda + k] : 0.0f;
        sA[i] = f2bf(v);
    }
    __syncthreads();

    const int wave = tid >> 5;
    const int lane = tid & 31;
    const int ntile = blockIdx.x * 4 + wave;
    const bool valid = (ntile * 16) < N;
    const int nCol = valid ? (ntile * 16 + (lane & 15)) : (lane & 15);
    const int row  = lane & 15;          // A row within tile (both lane halves)
    const int hs   = (lane >> 4) & 1;    // lane-half select

    v8f acc = {};
    for (int kb = 0; kb < K; kb += 32) {
        // A fragment: 16 consecutive K [k0..k0+15], k0 = kb + hs*16
        int k0 = kb + hs * 16;
        v8bf alo = *(const v8bf*)&sA[row * K + k0];
        v8bf ahi = *(const v8bf*)&sA[row * K + k0 + 8];
        v16bf afrag;
#pragma unroll
        for (int i = 0; i < 8; ++i) { afrag[i] = alo[i]; afrag[i + 8] = ahi[i]; }
        // B fragment: 16 consecutive K for fixed column n (contiguous row of W)
        int koff = kb + hs * 16;
        v16bf bfrag = *(const v16bf*)&W[(size_t)nCol * K + koff];
        acc = __builtin_amdgcn_wmma_f32_16x16x32_bf16(false, afrag, false, bfrag,
                                                      (short)0, acc, false, false);
    }

    if (valid) {
#pragma unroll
        for (int r = 0; r < 8; ++r) {
            int m = blockIdx.y * 16 + r + hs * 8;
            float v = acc[r];
            if (epilogue == 1) {
                float xv = v + bias[nCol];
                v = (xv > 20.0f) ? xv : logf(1.0f + __expf(xv));
            }
            C[(size_t)m * ldc + nCol] = v;
        }
    }
}

// ---------------- causal depthwise conv (width 2) + SiLU ----------------
__global__ void conv_silu_kernel(const float* __restrict__ xz, const float* __restrict__ cw,
                                 const float* __restrict__ cb, float* __restrict__ xc) {
    size_t i = (size_t)blockIdx.x * blockDim.x + threadIdx.x;
    if (i >= (size_t)M_TOK * DINNER) return;
    int d = (int)(i % DINNER);
    size_t m = i / DINNER;
    int l = (int)(m % LSEQ);
    float cur  = xz[m * (2 * DINNER) + d];
    float prev = (l > 0) ? xz[(m - 1) * (2 * DINNER) + d] : 0.0f;
    float v = cur * cw[d * 2 + 1] + prev * cw[d * 2 + 0] + cb[d];
    xc[i] = silu_f(v);
}

// ---------------- selective scan: 1 lane per (b, d, state); 4-lane butterfly reduce ----
__global__ void scan_kernel(const float* __restrict__ dt, const float* __restrict__ xc,
                            const float* __restrict__ proj, const float* __restrict__ xz,
                            const float* __restrict__ A_log, const float* __restrict__ Dp,
                            float* __restrict__ y) {
    int g = blockIdx.x * blockDim.x + threadIdx.x;     // 0 .. B*DINNER*4-1
    int c = g >> 2;                                    // channel (b,d)
    int s = g & 3;                                     // state index
    int b = c / DINNER;
    int d = c - b * DINNER;
    float A  = -__expf(A_log[d * DSTATE + s]);
    float Dd = Dp[d];
    float h = 0.0f;
    for (int l = 0; l < LSEQ; ++l) {
        size_t m = (size_t)b * LSEQ + l;
        float dtv = dt[m * DINNER + d];
        float xv  = xc[m * DINNER + d];
        float Bs  = proj[m * PROJ_LD + DTRANK + s];
        float Cs  = proj[m * PROJ_LD + DTRANK + DSTATE + s];
        h = __expf(dtv * A) * h + dtv * xv * Bs;
        float p = h * Cs;
        p += __shfl_xor(p, 1, 32);
        p += __shfl_xor(p, 2, 32);
        if (s == 0) {
            float zv = xz[m * (2 * DINNER) + DINNER + d];
            y[m * DINNER + d] = (p + xv * Dd) * silu_f(zv);
        }
    }
}

// ---------------- MLP score chain: 256 -> 64 -> 16 -> 4 -> 1, relu all ----------------
__global__ void mlp_score_kernel(const float* __restrict__ ym,
                                 const float* __restrict__ w1, const float* __restrict__ b1,
                                 const float* __restrict__ w2, const float* __restrict__ b2,
                                 const float* __restrict__ w3, const float* __restrict__ b3,
                                 const float* __restrict__ w4, const float* __restrict__ b4,
                                 float* __restrict__ sc) {
    int m = blockIdx.x * blockDim.x + threadIdx.x;
    if (m >= M_TOK) return;
    const float* row = ym + (size_t)m * DMODEL;
    float h1[64];
#pragma unroll
    for (int j = 0; j < 64; ++j) h1[j] = b1[j];
    for (int k = 0; k < DMODEL; ++k) {
        float xk = row[k];
#pragma unroll
        for (int j = 0; j < 64; ++j) h1[j] += xk * w1[j * DMODEL + k];
    }
#pragma unroll
    for (int j = 0; j < 64; ++j) h1[j] = fmaxf(h1[j], 0.0f);
    float h2[16];
#pragma unroll
    for (int j = 0; j < 16; ++j) h2[j] = b2[j];
    for (int k = 0; k < 64; ++k) {
        float xk = h1[k];
#pragma unroll
        for (int j = 0; j < 16; ++j) h2[j] += xk * w2[j * 64 + k];
    }
#pragma unroll
    for (int j = 0; j < 16; ++j) h2[j] = fmaxf(h2[j], 0.0f);
    float h3[4];
#pragma unroll
    for (int j = 0; j < 4; ++j) {
        float a = b3[j];
#pragma unroll
        for (int k = 0; k < 16; ++k) a += h2[k] * w3[j * 16 + k];
        h3[j] = fmaxf(a, 0.0f);
    }
    float sv = b4[0];
#pragma unroll
    for (int k = 0; k < 4; ++k) sv += h3[k] * w4[k];
    sc[m] = fmaxf(sv, 0.0f);
}

// ---------------- softmax stats per batch (max, sum of exp) ----------------
__global__ void softmax_stats_kernel(const float* __restrict__ sc, float* __restrict__ stats) {
    int b = blockIdx.x, tid = threadIdx.x;
    __shared__ float red[256];
    float mx = -3.402823e38f;
    for (int l = tid; l < LSEQ; l += 256) mx = fmaxf(mx, sc[b * LSEQ + l]);
    red[tid] = mx; __syncthreads();
    for (int s = 128; s > 0; s >>= 1) { if (tid < s) red[tid] = fmaxf(red[tid], red[tid + s]); __syncthreads(); }
    float bm = red[0]; __syncthreads();
    float se = 0.0f;
    for (int l = tid; l < LSEQ; l += 256) se += __expf(sc[b * LSEQ + l] - bm);
    red[tid] = se; __syncthreads();
    for (int s = 128; s > 0; s >>= 1) { if (tid < s) red[tid] += red[tid + s]; __syncthreads(); }
    if (tid == 0) { stats[b * 2] = bm; stats[b * 2 + 1] = red[0]; }
}

__global__ void zero_out_kernel(float* __restrict__ out, int n) {
    int i = blockIdx.x * blockDim.x + threadIdx.x;
    if (i < n) out[i] = 0.0f;
}

// ---------------- weighted pooling: out[b,d] += sum_l softmax(sc)[b,l] * x[b,l,d] ------
__global__ void weighted_sum_kernel(const float* __restrict__ sc, const float* __restrict__ stats,
                                    const float* __restrict__ x, float* __restrict__ out) {
    int chunk = blockIdx.x;                 // BATCH * (LSEQ/256)
    int b  = chunk / (LSEQ / 256);
    int t0 = (chunk - b * (LSEQ / 256)) * 256;
    int tid = threadIdx.x;
    __shared__ float a[256];
    {
        size_t m = (size_t)b * LSEQ + t0 + tid;
        a[tid] = __expf(sc[m] - stats[b * 2]) / stats[b * 2 + 1];
    }
    __syncthreads();
    float part = 0.0f;
    int d = tid;                            // DMODEL == 256 == blockDim
    const float* xb = x + ((size_t)b * LSEQ + t0) * DMODEL;
    for (int t = 0; t < 256; ++t) part += a[t] * xb[(size_t)t * DMODEL + d];
    atomicAdd(&out[b * DMODEL + d], part);
}

// ---------------- driver ----------------
extern "C" void kernel_launch(void* const* d_in, const int* in_sizes, int n_in,
                              void* d_out, int out_size, void* d_ws, size_t ws_size,
                              hipStream_t stream) {
    const float* x         = (const float*)d_in[0];
    const float* in_proj_w = (const float*)d_in[1];
    const float* conv_w    = (const float*)d_in[2];
    const float* conv_b    = (const float*)d_in[3];
    const float* x_proj_w  = (const float*)d_in[4];
    const float* dt_proj_w = (const float*)d_in[5];
    const float* dt_proj_b = (const float*)d_in[6];
    const float* A_log     = (const float*)d_in[7];
    const float* Dp        = (const float*)d_in[8];
    const float* out_proj_w= (const float*)d_in[9];
    const float* att_w1 = (const float*)d_in[10]; const float* att_b1 = (const float*)d_in[11];
    const float* att_w2 = (const float*)d_in[12]; const float* att_b2 = (const float*)d_in[13];
    const float* att_w3 = (const float*)d_in[14]; const float* att_b3 = (const float*)d_in[15];
    const float* att_w4 = (const float*)d_in[16]; const float* att_b4 = (const float*)d_in[17];
    float* out = (float*)d_out;

    // workspace carve-out
    char* w = (char*)d_ws;
    auto alloc = [&](size_t bytes) -> void* {
        void* p = (void*)w; w += (bytes + 255) & ~(size_t)255; return p;
    };
    __bf16* w_inproj  = (__bf16*)alloc((size_t)1024 * 256 * 2);
    __bf16* w_xproj   = (__bf16*)alloc((size_t)32 * DINNER * 2);      // pad 24 -> 32 rows
    __bf16* w_dtproj  = (__bf16*)alloc((size_t)DINNER * 32 * 2);      // pad K 16 -> 32
    __bf16* w_outproj = (__bf16*)alloc((size_t)DMODEL * DINNER * 2);
    float*  xz     = (float*)alloc((size_t)M_TOK * 2 * DINNER * 4);
    float*  xc     = (float*)alloc((size_t)M_TOK * DINNER * 4);
    float*  proj   = (float*)alloc((size_t)M_TOK * PROJ_LD * 4);
    float*  dt     = (float*)alloc((size_t)M_TOK * DINNER * 4);
    float*  ybuf   = (float*)alloc((size_t)M_TOK * DINNER * 4);
    float*  ymodel = (float*)alloc((size_t)M_TOK * DMODEL * 4);
    float*  scores = (float*)alloc((size_t)M_TOK * 4);
    float*  stats  = (float*)alloc(64);

    // 1) weight conversion / padding
    auto cvt = [&](const float* s, __bf16* d, int rs, int rd, int cs, int cd) {
        int tot = rd * cd;
        convert_pad_kernel<<<(tot + 255) / 256, 256, 0, stream>>>(s, d, rs, rd, cs, cd);
    };
    cvt(in_proj_w,  w_inproj,  1024, 1024, 256, 256);
    cvt(x_proj_w,   w_xproj,   24,   32,   DINNER, DINNER);
    cvt(dt_proj_w,  w_dtproj,  DINNER, DINNER, 16, 32);
    cvt(out_proj_w, w_outproj, DMODEL, DMODEL, DINNER, DINNER);

    // 2) GEMM1: xz = x @ in_proj_w.T   (M=65536, K=256, N=1024)
    {
        dim3 grid((1024 + 63) / 64, M_TOK / 16);
        gemm_bf16_wmma<<<grid, 128, 16 * 256 * 2, stream>>>(
            x, 256, 256, w_inproj, xz, 1024, 1024, 256, 0, nullptr);
    }
    // 3) conv + silu
    {
        size_t tot = (size_t)M_TOK * DINNER;
        conv_silu_kernel<<<(unsigned)((tot + 255) / 256), 256, 0, stream>>>(xz, conv_w, conv_b, xc);
    }
    // 4) GEMM2: proj = xc @ x_proj_w.T  (K=512, N=24 padded to 32, ldc=32)
    {
        dim3 grid(1, M_TOK / 16);
        gemm_bf16_wmma<<<grid, 128, 16 * DINNER * 2, stream>>>(
            xc, DINNER, DINNER, w_xproj, proj, PROJ_LD, 32, DINNER, 0, nullptr);
    }
    // 5) GEMM3: dt = softplus(proj[:, :16] @ dt_proj_w.T + dt_proj_b)  (K=16 padded to 32, N=512)
    {
        dim3 grid((DINNER + 63) / 64, M_TOK / 16);
        gemm_bf16_wmma<<<grid, 128, 16 * 32 * 2, stream>>>(
            proj, PROJ_LD, 16, w_dtproj, dt, DINNER, DINNER, 32, 1, dt_proj_b);
    }
    // 6) selective scan + gating
    {
        int threads = BATCH * DINNER * DSTATE;     // 16384
        scan_kernel<<<threads / 256, 256, 0, stream>>>(dt, xc, proj, xz, A_log, Dp, ybuf);
    }
    // 7) GEMM4: ymodel = y @ out_proj_w.T  (K=512, N=256)
    {
        dim3 grid((DMODEL + 63) / 64, M_TOK / 16);
        gemm_bf16_wmma<<<grid, 128, 16 * DINNER * 2, stream>>>(
            ybuf, DINNER, DINNER, w_outproj, ymodel, DMODEL, DMODEL, DINNER, 0, nullptr);
    }
    // 8) MLP scores
    mlp_score_kernel<<<M_TOK / 256, 256, 0, stream>>>(
        ymodel, att_w1, att_b1, att_w2, att_b2, att_w3, att_b3, att_w4, att_b4, scores);
    // 9) softmax stats per batch
    softmax_stats_kernel<<<BATCH, 256, 0, stream>>>(scores, stats);
    // 10) weighted pooling
    zero_out_kernel<<<(BATCH * DMODEL + 255) / 256, 256, 0, stream>>>(out, BATCH * DMODEL);
    weighted_sum_kernel<<<BATCH * (LSEQ / 256), 256, 0, stream>>>(scores, stats, x, out);
}
